// SubstitutionEmbedding_34239479284418
// MI455X (gfx1250) — matrix operand order repaired and checked by reading, not compiled, not run on hardware
//
#include <hip/hip_runtime.h>
#include <hip/hip_bf16.h>
#include <stdint.h>
#include <stddef.h>

// ---- problem constants (from reference) ----
#define BATCH   4
#define SEQ     36864
#define DIM     256
#define L0MAX   28672      // max last-layer tokens
#define L1MAX   20480      // max penultimate-layer tokens
#define T0N     3584       // L0MAX/8  rows of y0 per batch
#define T1N     5120       // L1MAX/4  rows of out per batch
#define KGEMM0  2048       // 8*256
#define KGEMM1  1024       // 4*256
#define NCHUNK  144        // SEQ/256
#define MAXDEP  6

// ---- trivial vector types (union-safe) ----
typedef unsigned int u32x4 __attribute__((ext_vector_type(4)));
typedef float        f32x4 __attribute__((ext_vector_type(4)));
typedef __bf16       v16bf __attribute__((ext_vector_type(16)));
typedef float        v8f   __attribute__((ext_vector_type(8)));

union FragBF { u32x4 q[2]; v16bf v; };

__device__ __forceinline__ unsigned short f2bf(float f) {
  unsigned u = __float_as_uint(f);
  u += 0x7FFFu + ((u >> 16) & 1u);        // round-to-nearest-even
  return (unsigned short)(u >> 16);
}

// ---------------- counting / ranking ----------------
__global__ void k_zero(int* len1) {
  if (threadIdx.x < BATCH) len1[threadIdx.x] = 0;
}

__global__ void k_count(const int* __restrict__ depth, const int* __restrict__ value,
                        int* __restrict__ len1, int* __restrict__ counts2) {
  int b = blockIdx.y, c = blockIdx.x, t = threadIdx.x;
  int s = c * 256 + t;
  int dep = depth[b * SEQ + s];
  int val = value[b * SEQ + s];
  int f1 = (dep == MAXDEP - 1) ? 1 : 0;
  int f2 = (f1 && val == 2) ? 1 : 0;
  __shared__ int sh1[256], sh2[256];
  sh1[t] = f1; sh2[t] = f2;
  __syncthreads();
  for (int st = 128; st > 0; st >>= 1) {
    if (t < st) { sh1[t] += sh1[t + st]; sh2[t] += sh2[t + st]; }
    __syncthreads();
  }
  if (t == 0) {
    atomicAdd(&len1[b], sh1[0]);
    counts2[b * NCHUNK + c] = sh2[0];
  }
}

__global__ void k_scan(const int* __restrict__ counts2, int* __restrict__ offsets2) {
  int b = blockIdx.x;
  if (threadIdx.x == 0) {
    int acc = 0;
    for (int c = 0; c < NCHUNK; ++c) {
      offsets2[b * NCHUNK + c] = acc;
      acc += counts2[b * NCHUNK + c];
    }
  }
}

__global__ void k_rank(const int* __restrict__ depth, const int* __restrict__ value,
                       const int* __restrict__ offsets2, int* __restrict__ pos2) {
  int b = blockIdx.y, c = blockIdx.x, t = threadIdx.x;
  int s = c * 256 + t;
  int dep = depth[b * SEQ + s];
  int val = value[b * SEQ + s];
  int f2 = (dep == MAXDEP - 1 && val == 2) ? 1 : 0;
  __shared__ int sh[256];
  sh[t] = f2;
  __syncthreads();
  for (int st = 1; st < 256; st <<= 1) {
    int add = (t >= st) ? sh[t - st] : 0;
    __syncthreads();
    sh[t] += add;
    __syncthreads();
  }
  if (f2) {
    int rank = sh[t] - 1;                       // exclusive rank within chunk
    pos2[b * 4096 + offsets2[b * NCHUNK + c] + rank] = s;
  }
}

// ---------------- embedding + ragged split (f32 -> bf16) ----------------
__global__ void k_embed(const int* __restrict__ value, const int* __restrict__ depth,
                        const int* __restrict__ position,
                        const float* __restrict__ Ev, const float* __restrict__ Ed,
                        const float* __restrict__ Ep, const int* __restrict__ len1,
                        unsigned short* __restrict__ X0, unsigned short* __restrict__ X1) {
  int ti   = threadIdx.x >> 4;                  // token within 16-token group
  int lane = threadIdx.x & 15;                  // 16-dim slice owner
  int gtok = blockIdx.x * 16 + ti;
  int b = gtok / SEQ;
  int s = gtok - b * SEQ;
  int idx = b * SEQ + s;
  int v   = value[idx];
  int dep = depth[idx];
  int p0 = position[idx * 3 + 0];
  int p1 = position[idx * 3 + 1];
  int p2 = position[idx * 3 + 2];
  const f32x4* Ev4 = (const f32x4*)Ev;
  const f32x4* Ed4 = (const f32x4*)Ed;
  const f32x4* Ep4 = (const f32x4*)Ep;
  int base = lane * 4;                          // f32x4 index within 256-dim row
  unsigned pk[8];
#pragma unroll
  for (int j = 0; j < 4; ++j) {
    f32x4 r = Ev4[v * 64 + base + j] + Ed4[dep * 64 + base + j] +
              Ep4[p0 * 64 + base + j] + Ep4[p1 * 64 + base + j] +
              Ep4[p2 * 64 + base + j];
    pk[2 * j + 0] = (unsigned)f2bf(r[0]) | ((unsigned)f2bf(r[1]) << 16);
    pk[2 * j + 1] = (unsigned)f2bf(r[2]) | ((unsigned)f2bf(r[3]) << 16);
  }
  int ln1 = len1[b];
  unsigned short* dst;
  if (dep == MAXDEP - 1)
    dst = X1 + ((size_t)b * L1MAX + s) * DIM;             // layer-1 tokens are s < len1
  else
    dst = X0 + ((size_t)b * L0MAX + (s - ln1)) * DIM;     // layer-0 tokens follow
  u32x4 w0 = {pk[0], pk[1], pk[2], pk[3]};
  u32x4 w1 = {pk[4], pk[5], pk[6], pk[7]};
  u32x4* d4 = (u32x4*)(dst + lane * 16);
  d4[0] = w0; d4[1] = w1;

  // zero the padded tails (each token owns at most one extra zero-row)
  u32x4 z = {0u, 0u, 0u, 0u};
  if (dep == MAXDEP) {
    if (s < L1MAX) {                                      // X1 tail rows are exactly s in [len1, L1)
      u32x4* zd = (u32x4*)(X1 + ((size_t)b * L1MAX + s) * DIM + lane * 16);
      zd[0] = z; zd[1] = z;
    }
  } else {
    int tail = ln1 - (SEQ - L0MAX);                       // = L0MAX - len0
    if (s < tail) {
      int row = (SEQ - ln1) + s;                          // len0 + s
      u32x4* zd = (u32x4*)(X0 + ((size_t)b * L0MAX + row) * DIM + lane * 16);
      zd[0] = z; zd[1] = z;
    }
  }
}

// ---------------- weight transpose to [N=256, K] bf16 ----------------
// Wt[o, kpos*256 + e] = W[o, e, kpos]
__global__ void k_wt(const float* __restrict__ W, unsigned short* __restrict__ Wt,
                     int Kk, int kc) {
  int idx = blockIdx.x * 256 + threadIdx.x;
  int o = idx / Kk;
  int r = idx - o * Kk;
  int kpos = r >> 8, e = r & 255;
  Wt[idx] = f2bf(W[((o << 8) + e) * kc + kpos]);
}

// ---------------- bf16 WMMA GEMM: C[M,256] = A[M,K] * Wt[256,K]^T + bias ----------------
// block = 256 threads = 8 waves; block tile 128x128; wave tile 64x32 (4x2 WMMA tiles).
// K-loop stages 8KB A-tile + 8KB B-tile per step through LDS via the gfx1250 async
// global->LDS path (ASYNCcnt), double-buffered so loads of step i+1 overlap the 8
// WMMAs of step i.
__global__ void __launch_bounds__(256) k_gemm(const unsigned short* __restrict__ A,
                                              const unsigned short* __restrict__ Wt,
                                              const float* __restrict__ bias,
                                              float* __restrict__ C, int Kk) {
  __shared__ __align__(64) unsigned char sh[2 * 16384];   // [buf][A 8KB | B 8KB]
  int t = threadIdx.x;
  int wid = t >> 5, lanep = t & 31;
  int rowBase = blockIdx.x * 128, colBase = blockIdx.y * 128;
  int wm = wid >> 2, wn = wid & 3;           // 2x4 wave grid
  int half = lanep >> 4, l = lanep & 15;

  // cooperative async-load geometry: each thread owns 16B segments t and t+256
  // of each 8KB tile (512 segs * 16B = 8KB moved per tile per step).
  int r0 = t >> 2, q0 = t & 3;
  int r1 = r0 + 64;                           // (t+256)>>2
  // wave-relative LDS byte address = low 32 bits of the generic pointer (ISA 10.2)
  uint32_t ldsBase = (uint32_t)(size_t)(void*)sh;
  uint32_t lA0 = ldsBase + (uint32_t)(r0 * 64 + q0 * 16);
  uint32_t lA1 = ldsBase + (uint32_t)(r1 * 64 + q0 * 16);
  uint32_t lB0 = lA0 + 8192u;
  uint32_t lB1 = lA1 + 8192u;
  uint32_t gA0 = (uint32_t)((size_t)(rowBase + r0) * Kk * 2 + q0 * 16);
  uint32_t gA1 = (uint32_t)((size_t)(rowBase + r1) * Kk * 2 + q0 * 16);
  uint32_t gB0 = (uint32_t)((size_t)(colBase + r0) * Kk * 2 + q0 * 16);
  uint32_t gB1 = (uint32_t)((size_t)(colBase + r1) * Kk * 2 + q0 * 16);

  auto issue_tiles = [&](uint32_t bufOff, uint32_t kb) {
    asm volatile("global_load_async_to_lds_b128 %0, %1, %2"
                 :: "v"(lA0 + bufOff), "v"(gA0 + kb), "s"(A) : "memory");
    asm volatile("global_load_async_to_lds_b128 %0, %1, %2"
                 :: "v"(lA1 + bufOff), "v"(gA1 + kb), "s"(A) : "memory");
    asm volatile("global_load_async_to_lds_b128 %0, %1, %2"
                 :: "v"(lB0 + bufOff), "v"(gB0 + kb), "s"(Wt) : "memory");
    asm volatile("global_load_async_to_lds_b128 %0, %1, %2"
                 :: "v"(lB1 + bufOff), "v"(gB1 + kb), "s"(Wt) : "memory");
  };

  v8f zero8 = {0.f, 0.f, 0.f, 0.f, 0.f, 0.f, 0.f, 0.f};
  v8f acc[4][2];
#pragma unroll
  for (int mt = 0; mt < 4; ++mt)
#pragma unroll
    for (int nt = 0; nt < 2; ++nt) acc[mt][nt] = zero8;

  int nIter = Kk >> 5;                        // K-steps of 32
  issue_tiles(0u, 0u);                        // prologue: fill buffer 0
  uint32_t kb = 0;                            // K byte offset (32 bf16 = 64B per step)
  for (int i = 0; i < nIter; ++i) {
    asm volatile("s_wait_asynccnt 0x0" ::: "memory");   // this wave's tile data in LDS
    __syncthreads();                                     // everyone's data visible
    uint32_t bufOff = (uint32_t)(i & 1) * 16384u;
    if (i + 1 < nIter) issue_tiles(16384u - bufOff, kb + 64u);
    if (i + 2 < nIter && t < 128) {           // warm L2 two slices ahead (global_prefetch_b8)
      __builtin_prefetch(A + (size_t)(rowBase + t) * Kk + (kb >> 1) + 64, 0, 0);
      __builtin_prefetch(Wt + (size_t)(colBase + t) * Kk + (kb >> 1) + 64, 0, 0);
    }

    const unsigned short* shA = (const unsigned short*)(sh + bufOff);
    const unsigned short* shB = (const unsigned short*)(sh + bufOff + 8192);
    FragBF af[4], bfr[2];
    // A 16x32 bf16 frag (ISA 7.12.2): lanes<16 -> K0-7 @+0, K16-23 @+32 ; lanes>=16 -> K8-15 @+16, K24-31 @+48
#pragma unroll
    for (int mt = 0; mt < 4; ++mt) {
      const u32x4* p = (const u32x4*)(shA + (wm * 64 + mt * 16 + l) * 32);
      af[mt].q[0] = p[half];
      af[mt].q[1] = p[2 + half];
    }
    // B 32x16 bf16 frag: lanes<16 -> K0-7 @+0, K16-23 @+32 ; lanes>=16 -> K8-15 @+16, K24-31 @+48
#pragma unroll
    for (int nt = 0; nt < 2; ++nt) {
      const u32x4* p = (const u32x4*)(shB + (wn * 32 + nt * 16 + l) * 32);
      bfr[nt].q[0] = p[half];
      bfr[nt].q[1] = p[2 + half];
    }
#pragma unroll
    for (int mt = 0; mt < 4; ++mt)
#pragma unroll
      for (int nt = 0; nt < 2; ++nt)
        acc[mt][nt] = __builtin_amdgcn_wmma_f32_16x16x32_bf16(
            false, af[mt].v, false, bfr[nt].v, (short)0, acc[mt][nt], false, false);
    kb += 64u;
  }

  float bv[2];
#pragma unroll
  for (int nt = 0; nt < 2; ++nt) bv[nt] = bias[colBase + wn * 32 + nt * 16 + l];
#pragma unroll
  for (int mt = 0; mt < 4; ++mt)
#pragma unroll
    for (int nt = 0; nt < 2; ++nt)
#pragma unroll
      for (int vv = 0; vv < 8; ++vv) {
        int row = rowBase + wm * 64 + mt * 16 + vv + half * 8;  // C layout per ISA 7.12.2
        int col = colBase + wn * 32 + nt * 16 + l;
        C[(size_t)row * 256 + col] = acc[mt][nt][vv] + bv[nt];
      }
}

// ---------------- substitution: X1[b, pos2[b][r]] = bf16(Y0[b, r]) ----------------
__global__ void k_subst(const float* __restrict__ Y0, const int* __restrict__ pos2,
                        const int* __restrict__ len1, unsigned short* __restrict__ X1) {
  int b = blockIdx.y, r = blockIdx.x;
  int ln1 = len1[b];
  int Mb = (SEQ - ln1) >> 3;                 // valid y0 rows == '2' count per batch
  if (r >= Mb) return;
  int s = pos2[b * 4096 + r];
  const f32x4* src = (const f32x4*)(Y0 + ((size_t)b * T0N + r) * DIM);
  unsigned short* dst = X1 + ((size_t)b * L1MAX + s) * DIM;
  int t = threadIdx.x;                       // 64 threads x 4 floats
  f32x4 f = src[t];
  unsigned lo = (unsigned)f2bf(f[0]) | ((unsigned)f2bf(f[1]) << 16);
  unsigned hi = (unsigned)f2bf(f[2]) | ((unsigned)f2bf(f[3]) << 16);
  unsigned* d = (unsigned*)dst;
  d[t * 2] = lo; d[t * 2 + 1] = hi;
}

// ---------------- launch ----------------
extern "C" void kernel_launch(void* const* d_in, const int* in_sizes, int n_in,
                              void* d_out, int out_size, void* d_ws, size_t ws_size,
                              hipStream_t stream) {
  (void)in_sizes; (void)n_in; (void)out_size; (void)ws_size;
  const int*   value    = (const int*)d_in[0];
  const int*   depth    = (const int*)d_in[1];
  const int*   position = (const int*)d_in[2];
  const float* E_val    = (const float*)d_in[3];
  const float* E_dep    = (const float*)d_in[4];
  const float* E_pos    = (const float*)d_in[5];
  const float* W0       = (const float*)d_in[6];
  const float* b0       = (const float*)d_in[7];
  const float* W1       = (const float*)d_in[8];
  const float* b1       = (const float*)d_in[9];

  char* ws = (char*)d_ws;
  int* len1     = (int*)(ws + 0);
  int* counts2  = (int*)(ws + 1024);
  int* offsets2 = (int*)(ws + 8192);
  int* pos2     = (int*)(ws + 16384);                       // BATCH*4096 ints = 64 KB
  unsigned short* Wt0 = (unsigned short*)(ws + (size_t)(1u << 20));   // 1 MB
  unsigned short* Wt1 = (unsigned short*)(ws + (size_t)(2u << 20));   // 0.5 MB
  unsigned short* X0  = (unsigned short*)(ws + (size_t)(4u << 20));   // 58.7 MB bf16
  unsigned short* X1  = (unsigned short*)(ws + (size_t)(64u << 20));  // 41.9 MB bf16
  float*          Y0  = (float*)(ws + (size_t)(112u << 20));          // 14.7 MB f32

  k_zero<<<1, 32, 0, stream>>>(len1);
  k_count<<<dim3(NCHUNK, BATCH), 256, 0, stream>>>(depth, value, len1, counts2);
  k_scan<<<BATCH, 32, 0, stream>>>(counts2, offsets2);
  k_rank<<<dim3(NCHUNK, BATCH), 256, 0, stream>>>(depth, value, offsets2, pos2);
  k_embed<<<(BATCH * SEQ) / 16, 256, 0, stream>>>(value, depth, position,
                                                  E_val, E_dep, E_pos, len1, X0, X1);
  k_wt<<<(256 * KGEMM0) / 256, 256, 0, stream>>>(W0, Wt0, KGEMM0, 8);
  k_wt<<<(256 * KGEMM1) / 256, 256, 0, stream>>>(W1, Wt1, KGEMM1, 4);
  // GEMM1: y0 = x0_blocks * W0^T + b0   (14336 x 2048 x 256)
  k_gemm<<<dim3((BATCH * T0N) / 128, 2), 256, 0, stream>>>(X0, Wt0, b0, Y0, KGEMM0);
  k_subst<<<dim3(4096, BATCH), 64, 0, stream>>>(Y0, pos2, len1, X1);
  // GEMM2: out = x1_blocks * W1^T + b1  (20480 x 1024 x 256)
  k_gemm<<<dim3((BATCH * T1N) / 128, 2), 256, 0, stream>>>(X1, Wt1, b1, (float*)d_out, KGEMM1);
}
